// GraphSAGE_54039278518913
// MI455X (gfx1250) — matrix operand order repaired and checked by reading, not compiled, not run on hardware
//
#include <hip/hip_runtime.h>
#include <hip/hip_bf16.h>
#include <math.h>

// ---------- WMMA types ----------
typedef __attribute__((ext_vector_type(16))) __bf16 v16bf;
typedef __attribute__((ext_vector_type(8)))  float  v8f;

union BF16Frag {
    v16bf v;
    unsigned short u[16];
};

// round-to-nearest-even f32 -> bf16 (bit pattern)
static __device__ __forceinline__ unsigned short f2bf(float f) {
    unsigned int x = __float_as_uint(f);
    unsigned int r = (x + 0x7FFFu + ((x >> 16) & 1u)) >> 16;
    return (unsigned short)r;
}

// ---------- generic zero ----------
__global__ void zero_f32_kernel(float* __restrict__ p, long long n) {
    long long i = (long long)blockIdx.x * blockDim.x + threadIdx.x;
    if (i < n) p[i] = 0.0f;
}

// ---------- per-destination edge counts (shared by all 3 layers) ----------
__global__ void count_kernel(const long long* __restrict__ dst,
                             float* __restrict__ cnt, int E) {
    int e = blockIdx.x * blockDim.x + threadIdx.x;
    if (e < E) atomicAdd(&cnt[(int)dst[e]], 1.0f);
}

// ---------- edge gather + scatter-add: s[dst] += feat[src]  ----------
__global__ void scatter_kernel(const float* __restrict__ feat,
                               const long long* __restrict__ src,
                               const long long* __restrict__ dst,
                               float* __restrict__ sbuf,
                               int E, int fshift) {
    long long idx = (long long)blockIdx.x * blockDim.x + threadIdx.x;
    long long total = (long long)E << fshift;
    if (idx >= total) return;
    int f = (int)(idx & ((1 << fshift) - 1));
    int e = (int)(idx >> fshift);
    int sn = (int)src[e];
    int dn = (int)dst[e];
    float v = feat[((size_t)sn << fshift) + f];
    atomicAdd(&sbuf[((size_t)dn << fshift) + f], v);
}

// ---------- pack [Wl;Wr] column-major bf16, pad Fout -> FoutPad ----------
__global__ void pack_weights_kernel(const float* __restrict__ Wl,
                                    const float* __restrict__ Wr,
                                    const float* __restrict__ b,
                                    unsigned short* __restrict__ wpack,
                                    float* __restrict__ bpad,
                                    int Fin, int Fout, int FoutPad) {
    int n = threadIdx.x;
    if (n >= FoutPad) return;
    int K = 2 * Fin;
    for (int k = 0; k < K; ++k) {
        float v = 0.0f;
        if (n < Fout)
            v = (k < Fin) ? Wl[k * Fout + n] : Wr[(k - Fin) * Fout + n];
        wpack[n * K + k] = f2bf(v);
    }
    bpad[n] = (n < Fout) ? b[n] : 0.0f;
}

// ---------- fused mean-div + concat GEMM via bf16 WMMA ----------
// out[M, FoutPad] = [s/max(cnt,1) | xin] @ wpack^T(col-major packed) + bias
// block = 256 threads = 8 wave32, 128 rows per block; each wave: 16-row strip.
__global__ void sage_gemm_kernel(const float* __restrict__ sbuf,
                                 const float* __restrict__ cnt,
                                 const float* __restrict__ xin,
                                 const unsigned short* __restrict__ wpack,
                                 const float* __restrict__ bpad,
                                 float* __restrict__ tmp,
                                 int N, int Fin, int K, int kshift, int FoutPad) {
    __shared__ unsigned short lA[128 * 128];   // 32 KB: 128 rows x K bf16
    __shared__ unsigned short lB[64 * 128];    // 16 KB: FoutPad cols x K bf16
    __shared__ float rc[128];

    const int tid = threadIdx.x;
    const int rowbase = blockIdx.x * 128;

    // reciprocal mean denominators
    if (tid < 128) {
        int gr = rowbase + tid;
        float c = (gr < N) ? cnt[gr] : 1.0f;
        rc[tid] = 1.0f / fmaxf(c, 1.0f);
    }
    // stage packed weights (already bf16, col-major [FoutPad][K])
    const int nb = FoutPad * K;
    for (int i = tid; i < nb; i += 256) lB[i] = wpack[i];
    __syncthreads();

    // stage A: [mean | x] rows, convert f32 -> bf16
    const int na = 128 << kshift;
    for (int i = tid; i < na; i += 256) {
        int r = i >> kshift;
        int k = i & (K - 1);
        int gr = rowbase + r;
        float v = 0.0f;
        if (gr < N) {
            v = (k < Fin) ? sbuf[(size_t)gr * Fin + k] * rc[r]
                          : xin[(size_t)gr * Fin + (k - Fin)];
        }
        lA[i] = f2bf(v);
    }
    __syncthreads();

    const int wv   = tid >> 5;        // wave id: strip of 16 rows
    const int lane = tid & 31;
    const int rloc = (lane & 15) + wv * 16;  // A row (lane-major within strip)
    const int kh   = lane >> 4;              // lane-half selector
    const int ncol = lane & 15;              // output column within tile
    const int ntiles = FoutPad >> 4;
    const int ktiles = K >> 5;

    for (int ft = 0; ft < ntiles; ++ft) {
        v8f acc = {};
        for (int kt = 0; kt < ktiles; ++kt) {
            BF16Frag af, bfr;
            const int kbase = kt * 32;
            const int arow = rloc << kshift;
            const int brow = ((ft * 16 + ncol) << kshift);
#pragma unroll
            for (int v = 0; v < 8; ++v) {
                // A 16x32 bf16 layout: V0..3 -> K = 2v + 8*kh ; V4..7 -> K = 16 + 2(v-4) + 8*kh
                int ka = kbase + ((v < 4) ? (2 * v + 8 * kh)
                                          : (16 + 2 * (v - 4) + 8 * kh));
                af.u[2 * v]     = lA[arow + ka];
                af.u[2 * v + 1] = lA[arow + ka + 1];
                // B 32x16 bf16 layout: same K mapping as A (V0..3 -> K = 2v + 8*kh ; V4..7 -> K = 16 + 2(v-4) + 8*kh)
                int kb = ka;
                bfr.u[2 * v]     = lB[brow + kb];
                bfr.u[2 * v + 1] = lB[brow + kb + 1];
            }
            acc = __builtin_amdgcn_wmma_f32_16x16x32_bf16(
                false, af.v, false, bfr.v, (short)0, acc, false, false);
        }
        const float bias = bpad[ft * 16 + ncol];
#pragma unroll
        for (int g = 0; g < 8; ++g) {
            int grow = rowbase + wv * 16 + g + 8 * kh;  // C layout: VGPR g -> M = g + 8*laneHalf
            if (grow < N)
                tmp[(size_t)grow * FoutPad + ft * 16 + ncol] = acc[g] + bias;
        }
    }
}

// ---------- row L2 normalize (+ optional ReLU) ----------
__global__ void norm_relu_kernel(const float* __restrict__ tmp,
                                 float* __restrict__ hout,
                                 int N, int FoutPad, int Fout, int do_relu) {
    int i = blockIdx.x * blockDim.x + threadIdx.x;
    if (i >= N) return;
    const float* row = tmp + (size_t)i * FoutPad;
    float ss = 0.0f;
    for (int j = 0; j < Fout; ++j) { float v = row[j]; ss += v * v; }
    float inv = 1.0f / fmaxf(sqrtf(ss), 1e-12f);
    float* orow = hout + (size_t)i * Fout;
    for (int j = 0; j < Fout; ++j) {
        float v = row[j] * inv;
        if (do_relu) v = fmaxf(v, 0.0f);
        orow[j] = v;
    }
}

// ---------- final layer: L2 normalize + log_softmax over 10 classes ----------
__global__ void final_kernel(const float* __restrict__ tmp,
                             float* __restrict__ out, int N) {
    int i = blockIdx.x * blockDim.x + threadIdx.x;
    if (i >= N) return;
    const float* row = tmp + (size_t)i * 16;
    float v[10];
    float ss = 0.0f;
    for (int j = 0; j < 10; ++j) { v[j] = row[j]; ss += v[j] * v[j]; }
    float inv = 1.0f / fmaxf(sqrtf(ss), 1e-12f);
    float m = -INFINITY;
    for (int j = 0; j < 10; ++j) { v[j] *= inv; m = fmaxf(m, v[j]); }
    float se = 0.0f;
    for (int j = 0; j < 10; ++j) se += expf(v[j] - m);
    float lse = m + logf(se);
    for (int j = 0; j < 10; ++j) out[(size_t)i * 10 + j] = v[j] - lse;
}

// =======================================================================
extern "C" void kernel_launch(void* const* d_in, const int* in_sizes, int n_in,
                              void* d_out, int out_size, void* d_ws, size_t ws_size,
                              hipStream_t stream) {
    const float*     x   = (const float*)d_in[0];
    const long long* ei  = (const long long*)d_in[1];
    const float*     W1l = (const float*)d_in[2];
    const float*     W1r = (const float*)d_in[3];
    const float*     b1  = (const float*)d_in[4];
    const float*     W2l = (const float*)d_in[5];
    const float*     W2r = (const float*)d_in[6];
    const float*     b2  = (const float*)d_in[7];
    const float*     W3l = (const float*)d_in[8];
    const float*     W3r = (const float*)d_in[9];
    const float*     b3  = (const float*)d_in[10];

    const int N = in_sizes[0] / 64;
    const int E = in_sizes[1] / 2;
    const long long* src = ei;
    const long long* dst = ei + E;

    // workspace layout
    float* sbuf = (float*)d_ws;                       // N*64
    float* cnt  = sbuf + (size_t)N * 64;              // N
    float* h1   = cnt + N;                            // N*64
    float* h2   = h1 + (size_t)N * 64;                // N*32
    float* tmp  = h2 + (size_t)N * 32;                // N*64
    unsigned short* wpack = (unsigned short*)(tmp + (size_t)N * 64); // 16384 bf16
    float* bpad = (float*)(wpack + 16384);            // 64

    const int T = 256;
    auto blks = [](long long n, int t) { return (unsigned int)((n + t - 1) / t); };

    // ---- counts (dst-only, reused for all layers) ----
    zero_f32_kernel<<<blks(N, T), T, 0, stream>>>(cnt, N);
    count_kernel<<<blks(E, T), T, 0, stream>>>(dst, cnt, E);

    // ---- layer 1: Fin=64, Fout=64 ----
    zero_f32_kernel<<<blks((long long)N * 64, T), T, 0, stream>>>(sbuf, (long long)N * 64);
    scatter_kernel<<<blks((long long)E << 6, T), T, 0, stream>>>(x, src, dst, sbuf, E, 6);
    pack_weights_kernel<<<1, 64, 0, stream>>>(W1l, W1r, b1, wpack, bpad, 64, 64, 64);
    sage_gemm_kernel<<<blks(N, 128), T, 0, stream>>>(sbuf, cnt, x, wpack, bpad, tmp,
                                                     N, 64, 128, 7, 64);
    norm_relu_kernel<<<blks(N, T), T, 0, stream>>>(tmp, h1, N, 64, 64, 1);

    // ---- layer 2: Fin=64, Fout=32 ----
    zero_f32_kernel<<<blks((long long)N * 64, T), T, 0, stream>>>(sbuf, (long long)N * 64);
    scatter_kernel<<<blks((long long)E << 6, T), T, 0, stream>>>(h1, src, dst, sbuf, E, 6);
    pack_weights_kernel<<<1, 32, 0, stream>>>(W2l, W2r, b2, wpack, bpad, 64, 32, 32);
    sage_gemm_kernel<<<blks(N, 128), T, 0, stream>>>(sbuf, cnt, h1, wpack, bpad, tmp,
                                                     N, 64, 128, 7, 32);
    norm_relu_kernel<<<blks(N, T), T, 0, stream>>>(tmp, h2, N, 32, 32, 1);

    // ---- layer 3: Fin=32, Fout=10 (pad 16) ----
    zero_f32_kernel<<<blks((long long)N * 32, T), T, 0, stream>>>(sbuf, (long long)N * 32);
    scatter_kernel<<<blks((long long)E << 5, T), T, 0, stream>>>(h2, src, dst, sbuf, E, 5);
    pack_weights_kernel<<<1, 16, 0, stream>>>(W3l, W3r, b3, wpack, bpad, 32, 10, 16);
    sage_gemm_kernel<<<blks(N, 128), T, 0, stream>>>(sbuf, cnt, h2, wpack, bpad, tmp,
                                                     N, 32, 64, 6, 16);
    final_kernel<<<blks(N, T), T, 0, stream>>>(tmp, (float*)d_out, N);
}
